// Down4SparseMiddleNetwork_1425929142958
// MI455X (gfx1250) — matrix operand order repaired and checked by reading, not compiled, not run on hardware
//
#include <hip/hip_runtime.h>
#include <cstdint>
#include <cstddef>

typedef __attribute__((ext_vector_type(16))) _Float16 v16h;
typedef __attribute__((ext_vector_type(8)))  float    v8f;

union Frag { uint4 u[2]; v16h h; };

// ---------------- utility kernels ----------------

__global__ void zero_kernel(uint32_t* p, long n) {
  long i = (long)blockIdx.x * blockDim.x + threadIdx.x;
  if (i < n) p[i] = 0u;
}

// fold BN into scale/bias: y = x*scale + bias
__global__ void bn_prep_kernel(const float* gamma, const float* beta, const float* mean,
                               const float* var, float* scale, float* bias, int n) {
  int i = blockIdx.x * blockDim.x + threadIdx.x;
  if (i >= n) return;
  float s = gamma[i] * rsqrtf(var[i] + 1e-3f);
  scale[i] = s;
  bias[i]  = beta[i] - mean[i] * s;
}

// Pack fp32 (3,3,3,CIN,COUT) weights into f16 WMMA B-fragments.
// Fragment (t, kc, nt): 32 lanes x 16 f16, lane holds col n = nt*16 + (lane&15),
// K = kc*32 + (lane>=16 ? 16 : 0) + e  (e = 0..15), stored lane-major contiguous.
__global__ void pack_weights_kernel(const float* w, _Float16* wpk, int CIN, int COUT) {
  int idx = blockIdx.x * blockDim.x + threadIdx.x;
  int total = 27 * CIN * COUT;
  if (idx >= total) return;
  int e    = idx & 15;
  int lane = (idx >> 4) & 31;
  int frag = idx >> 9;
  int NW = COUT >> 4;
  int KC = CIN >> 5;
  int nt = frag % NW;
  int kc = (frag / NW) % KC;
  int t  = frag / (NW * KC);
  int k  = kc * 32 + ((lane >= 16) ? 16 : 0) + e;
  int n  = nt * 16 + (lane & 15);
  wpk[idx] = (_Float16)w[(t * CIN + k) * COUT + n];
}

__global__ void scatter_kernel(const float* vf, const int* coors, _Float16* g,
                               unsigned char* occ, int nvox, int H, int W) {
  int i = blockIdx.x * blockDim.x + threadIdx.x;
  if (i >= nvox * 32) return;
  int v = i >> 5, c = i & 31;
  int z = coors[v*4+1], y = coors[v*4+2], x = coors[v*4+3];
  long lin = ((long)z * H + y) * W + x;
  g[lin * 32 + c] = (_Float16)vf[i];
  if (c == 0) occ[lin] = 1;
}

// occ_out = reduce_window(occ_in, 3x3x3, stride, pad 1) > 0
__global__ void occ_down_kernel(const unsigned char* oi, unsigned char* oo,
                                int Di,int Hi,int Wi,int Do,int Ho,int Wo,
                                int sz,int sy,int sx) {
  long i = (long)blockIdx.x*blockDim.x + threadIdx.x;
  long tot = (long)Do*Ho*Wo;
  if (i >= tot) return;
  int x = (int)(i % Wo); int y = (int)((i / Wo) % Ho); int z = (int)(i / ((long)Wo*Ho));
  int r = 0;
  for (int dz=-1; dz<=1; ++dz)
    for (int dy=-1; dy<=1; ++dy)
      for (int dx=-1; dx<=1; ++dx) {
        int zi=z*sz+dz, yi=y*sy+dy, xi=x*sx+dx;
        if (zi>=0&&zi<Di&&yi>=0&&yi<Hi&&xi>=0&&xi<Wi) r |= oi[((long)zi*Hi+yi)*Wi+xi];
      }
  oo[i] = r ? 1 : 0;
}

// NDHWC f16 grid -> (C*D, H, W) f32 dense output
__global__ void dense_out_kernel(const _Float16* g, float* out, int DD,int HH,int WW,int C) {
  long i = (long)blockIdx.x*blockDim.x + threadIdx.x;
  long tot = (long)C*DD*HH*WW;
  if (i >= tot) return;
  int x = (int)(i % WW);
  int y = (int)((i / WW) % HH);
  int z = (int)((i / ((long)WW*HH)) % DD);
  int c = (int)(i / ((long)WW*HH*DD));
  out[i] = (float)g[(((long)z*HH + y)*WW + x)*C + c];
}

// ---------------- implicit-GEMM 3x3x3 conv + BN + ReLU + occupancy mask ----------------
// Block covers 32 output voxels x COUT channels. Each of the COUT/16 waves computes
// two 16x16 tiles (M-blocked) sharing one B fragment -> 2 WMMA per weight fetch.
// A tiles (32 voxels x CIN f16) are double-buffered in LDS: tap t+1 streams in
// while tap t feeds the WMMAs; one barrier per tap.

template<int CIN, int COUT, int SZ, int SY, int SX>
__launch_bounds__(COUT*2)
__global__ void conv3x3x3_wmma(const _Float16* __restrict__ in,
                               const _Float16* __restrict__ wpk,
                               const float* __restrict__ scale,
                               const float* __restrict__ bias,
                               const unsigned char* __restrict__ occ,
                               _Float16* __restrict__ out,
                               int Din,int Hin,int Win,
                               int Dout,int Hout,int Wout) {
  constexpr int NW   = COUT / 16;   // waves per block = column tiles
  constexpr int KC   = CIN / 32;    // K-chunks per tap
  constexpr int CH8  = CIN / 8;     // uint4 chunks per A row (power of two)
  constexpr int NCHUNK = 32 * CH8;  // chunks per 32-voxel tile
  constexpr int NTHR = COUT * 2;
  __shared__ _Float16 sA[2][32 * CIN];

  const int tid  = threadIdx.x;
  const int lane = tid & 31;
  const int wave = tid >> 5;
  const int xt = blockIdx.x << 5;   // 32-voxel x-tile
  const int yo = blockIdx.y;
  const int zo = blockIdx.z;

  // A-fragment addressing (this code's contiguous K order, same as B packing):
  // lane<16 holds row M=lane, K {0..15}; lane>=16 holds M=lane-16, K {16..31}
  const int mrow  = lane & 15;
  const int kbase = (lane < 16) ? 0 : 16;

  v8f acc0 = {};
  v8f acc1 = {};

  // cooperative halo-checked global -> LDS load of 32 x CIN f16
  auto loadTile = [&](int dz, int dy, int dx, int buf) {
    const int zi = zo*SZ + dz, yi = yo*SY + dy;
    const bool zyok = (zi>=0) && (zi<Din) && (yi>=0) && (yi<Hin);
    for (int q = tid; q < NCHUNK; q += NTHR) {
      const int m = q / CH8, c8 = q % CH8;        // shifts/masks (CH8 pow2)
      const int xi = (xt + m)*SX + dx;
      uint4 val = make_uint4(0u,0u,0u,0u);
      if (zyok && xi >= 0 && xi < Win)
        val = reinterpret_cast<const uint4*>(in + (((long)zi*Hin + yi)*Win + xi)*CIN)[c8];
      reinterpret_cast<uint4*>(sA[buf])[q] = val;
    }
  };

  // prologue: stage tap 0
  loadTile(-1, -1, -1, 0);
  __syncthreads();

  int dx = -1, dy = -1, dz = -1;
  for (int t = 0; t < 27; ++t) {
    // next tap coordinates (incremental, no div/mod)
    int ndx = dx + 1, ndy = dy, ndz = dz;
    if (ndx == 2) { ndx = -1; ndy = dy + 1; if (ndy == 2) { ndy = -1; ndz = dz + 1; } }

    if (t < 26) {
      loadTile(ndz, ndy, ndx, (t + 1) & 1);   // stream tap t+1 into other buffer
      __builtin_prefetch(wpk + ((size_t)(t+1)*KC*NW + wave)*512 + lane*16, 0, 1);
    }

    // consume tap t
    const _Float16* wbase = wpk + ((size_t)t*KC*NW + wave)*512;
    const _Float16* abase = sA[t & 1];
#pragma unroll
    for (int kc = 0; kc < KC; ++kc) {
      Frag a0, a1, b;
      const uint4* ap0 = reinterpret_cast<const uint4*>(abase + mrow*CIN + kc*32 + kbase);
      const uint4* ap1 = reinterpret_cast<const uint4*>(abase + (16 + mrow)*CIN + kc*32 + kbase);
      a0.u[0] = ap0[0];  a0.u[1] = ap0[1];     // K kbase..+15
      a1.u[0] = ap1[0];  a1.u[1] = ap1[1];
      const uint4* bp = reinterpret_cast<const uint4*>(wbase + (size_t)kc*NW*512 + lane*16);
      b.u[0] = bp[0];    b.u[1] = bp[1];
      acc0 = __builtin_amdgcn_wmma_f32_16x16x32_f16(false, a0.h, false, b.h,
                                                    (short)0, acc0, false, false);
      acc1 = __builtin_amdgcn_wmma_f32_16x16x32_f16(false, a1.h, false, b.h,
                                                    (short)0, acc1, false, false);
    }
    __syncthreads();   // writes of tap t+1 visible; reads of tap t done before t+2 overwrites
    dx = ndx; dy = ndy; dz = ndz;
  }

  // epilogue: BN (folded), occupancy mask, ReLU, f16 store
  // C layout: lane<16 -> M = i, lane>=16 -> M = 8+i ; N = lane&15
  const int n  = (wave << 4) + (lane & 15);
  const float sc = scale[n], bi = bias[n];
  const int mbase = (lane < 16) ? 0 : 8;
#pragma unroll
  for (int i = 0; i < 8; ++i) {
    const int xo0 = xt + mbase + i;
    if (xo0 < Wout) {
      const long vox = ((long)zo*Hout + yo)*Wout + xo0;
      float yv = acc0[i]*sc + bi;
      yv = occ[vox] ? fmaxf(yv, 0.0f) : 0.0f;
      out[vox*COUT + n] = (_Float16)yv;
    }
    const int xo1 = xt + 16 + mbase + i;
    if (xo1 < Wout) {
      const long vox = ((long)zo*Hout + yo)*Wout + xo1;
      float yv = acc1[i]*sc + bi;
      yv = occ[vox] ? fmaxf(yv, 0.0f) : 0.0f;
      out[vox*COUT + n] = (_Float16)yv;
    }
  }
}

// ---------------- host launcher ----------------

extern "C" void kernel_launch(void* const* d_in, const int* in_sizes, int n_in,
                              void* d_out, int out_size, void* d_ws, size_t ws_size,
                              hipStream_t stream) {
  (void)in_sizes; (void)n_in; (void)out_size; (void)ws_size;
  // JAX pytree leaf order (dict keys sorted):
  // 0: batch_size, 1: coors,
  // 2..6:  c1  (beta,gamma,mean,var,w)
  // 7..11: c2, 12..16: c3,
  // 17..21: s1[0], 22..26: s1[1],
  // 27..31: s2[0], 32..36: s2[1],
  // 37..41: s3[0], 42..46: s3[1],
  // 47: voxel_features
  const int*   coors = (const int*)d_in[1];
  const float* vf    = (const float*)d_in[47];

  struct LCfg { int base, cin, cout; };
  const LCfg L[9] = {
    {17, 32, 64},   // s1[0]
    {22, 64, 64},   // s1[1]
    {2,  64, 64},   // c1 (stride 2)
    {27, 64, 128},  // s2[0]
    {32, 128,128},  // s2[1]
    {7,  128,128},  // c2 (stride 2)
    {37, 128,128},  // s3[0]
    {42, 128,128},  // s3[1]
    {12, 128,128},  // c3 (stride 1,2,2)
  };

  char* ws = (char*)d_ws;
  size_t off = 0;
  auto alloc = [&](size_t bytes) -> char* {
    char* p = ws + off;
    off = (off + bytes + 255) & ~(size_t)255;
    return p;
  };

  // ping-pong feature grids (max: 16*160*160 voxels * 64 ch * 2B = 52.4 MB)
  _Float16* bufA = (_Float16*)alloc(52428800);
  _Float16* bufB = (_Float16*)alloc(52428800);
  unsigned char* occ0 = (unsigned char*)alloc(409600);  // 16x160x160
  unsigned char* occ1 = (unsigned char*)alloc(51200);   // 8x80x80
  unsigned char* occ2 = (unsigned char*)alloc(6400);    // 4x40x40
  unsigned char* occ3 = (unsigned char*)alloc(1600);    // 4x20x20

  _Float16* wpk[9]; float* sc[9]; float* bi[9];
  for (int l = 0; l < 9; ++l) {
    size_t welem = (size_t)27 * L[l].cin * L[l].cout;
    wpk[l] = (_Float16*)alloc(welem * 2);
    sc[l]  = (float*)alloc((size_t)L[l].cout * 4);
    bi[l]  = (float*)alloc((size_t)L[l].cout * 4);
  }

  // weight packing + BN folding (deterministic, every call)
  for (int l = 0; l < 9; ++l) {
    int b = L[l].base;
    int welem = 27 * L[l].cin * L[l].cout;
    pack_weights_kernel<<<(welem + 255)/256, 256, 0, stream>>>(
        (const float*)d_in[b+4], wpk[l], L[l].cin, L[l].cout);
    bn_prep_kernel<<<1, L[l].cout, 0, stream>>>(
        (const float*)d_in[b+1], (const float*)d_in[b+0],
        (const float*)d_in[b+2], (const float*)d_in[b+3],
        sc[l], bi[l], L[l].cout);
  }

  // zero input grid (16x160x160x32 f16 = 6.55M dwords) + occupancy, then scatter
  zero_kernel<<<(6553600 + 255)/256, 256, 0, stream>>>((uint32_t*)bufA, 6553600);
  zero_kernel<<<(102400 + 255)/256, 256, 0, stream>>>((uint32_t*)occ0, 102400);
  scatter_kernel<<<(60000*32 + 255)/256, 256, 0, stream>>>(vf, coors, bufA, occ0, 60000, 160, 160);

  // s1: 32->64, 64->64 @ 16x160x160   (x-tiles of 32: 160/32 = 5)
  conv3x3x3_wmma<32,64,1,1,1><<<dim3(5,160,16), 128, 0, stream>>>(
      bufA, wpk[0], sc[0], bi[0], occ0, bufB, 16,160,160, 16,160,160);
  conv3x3x3_wmma<64,64,1,1,1><<<dim3(5,160,16), 128, 0, stream>>>(
      bufB, wpk[1], sc[1], bi[1], occ0, bufA, 16,160,160, 16,160,160);

  // c1: stride 2 -> 8x80x80   (ceil(80/32) = 3)
  occ_down_kernel<<<(51200+255)/256, 256, 0, stream>>>(occ0, occ1, 16,160,160, 8,80,80, 2,2,2);
  conv3x3x3_wmma<64,64,2,2,2><<<dim3(3,80,8), 128, 0, stream>>>(
      bufA, wpk[2], sc[2], bi[2], occ1, bufB, 16,160,160, 8,80,80);

  // s2: 64->128, 128->128 @ 8x80x80
  conv3x3x3_wmma<64,128,1,1,1><<<dim3(3,80,8), 256, 0, stream>>>(
      bufB, wpk[3], sc[3], bi[3], occ1, bufA, 8,80,80, 8,80,80);
  conv3x3x3_wmma<128,128,1,1,1><<<dim3(3,80,8), 256, 0, stream>>>(
      bufA, wpk[4], sc[4], bi[4], occ1, bufB, 8,80,80, 8,80,80);

  // c2: stride 2 -> 4x40x40   (ceil(40/32) = 2)
  occ_down_kernel<<<(6400+255)/256, 256, 0, stream>>>(occ1, occ2, 8,80,80, 4,40,40, 2,2,2);
  conv3x3x3_wmma<128,128,2,2,2><<<dim3(2,40,4), 256, 0, stream>>>(
      bufB, wpk[5], sc[5], bi[5], occ2, bufA, 8,80,80, 4,40,40);

  // dense1 = c2 output (launched before s3 can overwrite bufA; stream-ordered)
  dense_out_kernel<<<(819200+255)/256, 256, 0, stream>>>(bufA, (float*)d_out, 4,40,40,128);

  // s3: 128->128 x2 @ 4x40x40
  conv3x3x3_wmma<128,128,1,1,1><<<dim3(2,40,4), 256, 0, stream>>>(
      bufA, wpk[6], sc[6], bi[6], occ2, bufB, 4,40,40, 4,40,40);
  conv3x3x3_wmma<128,128,1,1,1><<<dim3(2,40,4), 256, 0, stream>>>(
      bufB, wpk[7], sc[7], bi[7], occ2, bufA, 4,40,40, 4,40,40);

  // c3: stride (1,2,2) -> 4x20x20   (ceil(20/32) = 1)
  occ_down_kernel<<<(1600+255)/256, 256, 0, stream>>>(occ2, occ3, 4,40,40, 4,20,20, 1,2,2);
  conv3x3x3_wmma<128,128,1,2,2><<<dim3(1,20,4), 256, 0, stream>>>(
      bufA, wpk[8], sc[8], bi[8], occ3, bufB, 4,40,40, 4,20,20);

  dense_out_kernel<<<(204800+255)/256, 256, 0, stream>>>(bufB, (float*)d_out + 819200, 4,20,20,128);
}
